// Atten_Head_83863531422422
// MI455X (gfx1250) — matrix operand order repaired and checked by reading, not compiled, not run on hardware
//
#include <hip/hip_runtime.h>
#include <hip/hip_bf16.h>

// ---------------------------------------------------------------------------
// Types for CDNA5 WMMA (wave32, 16x16x32 bf16 -> f32) and TDM descriptors
// ---------------------------------------------------------------------------
typedef __attribute__((ext_vector_type(16))) __bf16 v16bf;
typedef __attribute__((ext_vector_type(8)))  __bf16 bf16x8;
typedef __attribute__((ext_vector_type(8)))  float  v8f;
typedef __attribute__((ext_vector_type(4)))  unsigned int u32x4;
typedef __attribute__((ext_vector_type(8)))  int  i32x8;
typedef __attribute__((ext_vector_type(4)))  int  i32x4;

#define B_   8
#define S_   2048
#define D_   1024

#define MB   128   // block tile M (8 waves x 16 rows)
#define NB   64    // block tile N (4 x 16 frags per wave)
#define KB   32    // k step per WMMA
#define LDSP 40    // padded LDS row stride in halves (32 + 8, keeps 16B align)

static __device__ __forceinline__ v16bf join16(bf16x8 lo, bf16x8 hi) {
    return __builtin_shufflevector(lo, hi, 0,1,2,3,4,5,6,7,8,9,10,11,12,13,14,15);
}

// A fragment (16x32 bf16): lane row = lr, chunks at k = hi*8 and hi*8+16
static __device__ __forceinline__ v16bf load_a_frag(const __bf16* rowp, int hi) {
    bf16x8 c0 = *(const bf16x8*)(rowp + hi * 8);
    bf16x8 c1 = *(const bf16x8*)(rowp + hi * 8 + 16);
    return join16(c0, c1);
}
// B fragment (32x16 bf16): lane col (= row of Bt) = lr, chunks at k = hi*8 and hi*8+16
static __device__ __forceinline__ v16bf load_b_frag(const __bf16* rowp, int hi) {
    bf16x8 c0 = *(const bf16x8*)(rowp + hi * 8);
    bf16x8 c1 = *(const bf16x8*)(rowp + hi * 8 + 16);
    return join16(c0, c1);
}

// ---------------------------------------------------------------------------
// Tensor Data Mover: DMA a [rows x 32] bf16 tile (row stride = rowlen elems)
// from global to LDS, inserting 8 halves of padding per row so the LDS tile
// has stride LDSP=40 halves (pad: 4 DWORDs after every 16 DWORDs).
// D# bit layout per CDNA5 ISA.
// ---------------------------------------------------------------------------
static __device__ __forceinline__ void tdm_load_tile(
    const __bf16* gsrc, __bf16* ldst, int rows, int rowlen_elems)
{
    const unsigned long long ga = (unsigned long long)(const void*)gsrc;
    const unsigned int lds_off  = (unsigned int)(unsigned long long)(void*)ldst;

    u32x4 g0;
    g0[0] = 1u;                                   // count=1 (valid), user mode
    g0[1] = lds_off;                              // lds_addr (bytes)
    g0[2] = (unsigned int)ga;                     // global_addr[31:0]
    g0[3] = ((unsigned int)(ga >> 32) & 0x1FFFFFFu) | (2u << 30);  // addr[56:32] | type=2

    i32x8 g1;
    g1[0] = (1 << 16)                             // data_size = 2 bytes
          | (1 << 20)                             // pad_enable
          | (3 << 22)                             // pad_interval: every 16 DWORDs (64B row)
          | (3 << 25);                            // pad_amount: 4 DWORDs (8 halves)
    g1[1] = (rowlen_elems & 0xFFFF) << 16;        // tensor_dim0[15:0]
    g1[2] = ((rowlen_elems >> 16) & 0xFFFF)       // tensor_dim0[31:16]
          | ((rows & 0xFFFF) << 16);              // tensor_dim1[15:0]
    g1[3] = (KB << 16);                           // tensor_dim1[31:16]=0 | tile_dim0=32
    g1[4] = rows & 0xFFFF;                        // tile_dim1=rows, tile_dim2=0
    g1[5] = rowlen_elems;                         // tensor_dim0_stride[31:0]
    g1[6] = 0;                                    // stride0 hi | stride1 lo
    g1[7] = 0;                                    // stride1 hi

    const i32x4 z4 = {0, 0, 0, 0};
#if defined(__clang_major__) && (__clang_major__ >= 23)
    const i32x8 z8 = {0, 0, 0, 0, 0, 0, 0, 0};
    __builtin_amdgcn_tensor_load_to_lds(g0, g1, z4, z4, z8, 0);
#else
    __builtin_amdgcn_tensor_load_to_lds(g0, g1, z4, z4, 0);
#endif
}

// ---------------------------------------------------------------------------
// Kernel 1: transpose-convert the three weight matrices W[d,e] -> Wt[e,d] bf16
// ---------------------------------------------------------------------------
__global__ __launch_bounds__(256) void wt_convert_kernel(
    const float* __restrict__ WQ, const float* __restrict__ WK,
    const float* __restrict__ WV,
    __bf16* __restrict__ WtQ, __bf16* __restrict__ WtK, __bf16* __restrict__ WtV)
{
    __shared__ __bf16 t[32][33];
    const float* W; __bf16* Wt;
    switch (blockIdx.z) {
        case 0:  W = WQ; Wt = WtQ; break;
        case 1:  W = WK; Wt = WtK; break;
        default: W = WV; Wt = WtV; break;
    }
    const int e0 = blockIdx.x * 32;
    const int d0 = blockIdx.y * 32;
    const int tx = threadIdx.x & 31;
    const int ty = threadIdx.x >> 5;   // 0..7
#pragma unroll
    for (int i = 0; i < 4; ++i) {
        const int dr = ty * 4 + i;
        t[dr][tx] = (__bf16)W[(size_t)(d0 + dr) * D_ + (e0 + tx)];
    }
    __syncthreads();
#pragma unroll
    for (int i = 0; i < 4; ++i) {
        const int er = ty * 4 + i;
        Wt[(size_t)(e0 + er) * D_ + (d0 + tx)] = t[tx][er];
    }
}

// ---------------------------------------------------------------------------
// Kernel 2: fused projection GEMM (f32 inputs converted to bf16 on the fly).
//   z=0: Qb[bs,e] = (X_q @ W_Q) * 1/sqrt(D)
//   z=1: Kb[bs,e] =  X_k @ W_K
//   z=2: Vt[b,e,s] = (X_v @ W_V)^T
// ---------------------------------------------------------------------------
__global__ __launch_bounds__(256) void proj_kernel(
    const float* __restrict__ Xq, const float* __restrict__ Xk,
    const float* __restrict__ Xv,
    const __bf16* __restrict__ WtQ, const __bf16* __restrict__ WtK,
    const __bf16* __restrict__ WtV,
    __bf16* __restrict__ Qb, __bf16* __restrict__ Kb, __bf16* __restrict__ Vt)
{
    __shared__ __align__(16) __bf16 lA[MB * LDSP];
    __shared__ __align__(16) __bf16 lB[NB * LDSP];

    const int z = blockIdx.z;
    const float*  X  = (z == 0) ? Xq  : (z == 1) ? Xk  : Xv;
    const __bf16* Wt = (z == 0) ? WtQ : (z == 1) ? WtK : WtV;

    const int m0 = blockIdx.y * MB;
    const int n0 = blockIdx.x * NB;
    const int tid  = threadIdx.x;
    const int wave = tid >> 5;
    const int lane = tid & 31;
    const int lr   = lane & 15;
    const int hi   = lane >> 4;

    v8f acc[4] = {};

    const int arow = tid >> 1, aseg = tid & 1;   // A tile: 128 rows x 2 segs of 16
    const int brow = tid >> 2, bseg = tid & 3;   // B tile: 64 rows x 4 segs of 8

    for (int k0 = 0; k0 < D_; k0 += KB) {
        {
            const float* src = X + (size_t)(m0 + arow) * D_ + k0 + aseg * 16;
            const float4* s4 = (const float4*)src;
            float f[16];
            *(float4*)(f + 0)  = s4[0];
            *(float4*)(f + 4)  = s4[1];
            *(float4*)(f + 8)  = s4[2];
            *(float4*)(f + 12) = s4[3];
            bf16x8 h0, h1;
#pragma unroll
            for (int i = 0; i < 8; ++i) { h0[i] = (__bf16)f[i]; h1[i] = (__bf16)f[8 + i]; }
            __bf16* dst = lA + arow * LDSP + aseg * 16;
            *(bf16x8*)(dst)     = h0;
            *(bf16x8*)(dst + 8) = h1;
        }
        {
            const __bf16* src = Wt + (size_t)(n0 + brow) * D_ + k0 + bseg * 8;
            *(bf16x8*)(lB + brow * LDSP + bseg * 8) = *(const bf16x8*)src;
        }
        if (k0 + KB < D_) {
            __builtin_prefetch(X + (size_t)(m0 + arow) * D_ + k0 + KB, 0, 1);
            __builtin_prefetch(Wt + (size_t)(n0 + brow) * D_ + k0 + KB, 0, 1);
        }
        __syncthreads();

        const v16bf a = load_a_frag(lA + (wave * 16 + lr) * LDSP, hi);
#pragma unroll
        for (int n = 0; n < 4; ++n) {
            const v16bf b = load_b_frag(lB + (n * 16 + lr) * LDSP, hi);
            acc[n] = __builtin_amdgcn_wmma_f32_16x16x32_bf16(
                false, a, false, b, (short)0, acc[n], false, false);
        }
        __syncthreads();
    }

    const float scale = (z == 0) ? 0.03125f : 1.0f;   // 1/sqrt(1024) folded into Q
    if (z < 2) {
        __bf16* out = (z == 0) ? Qb : Kb;
#pragma unroll
        for (int n = 0; n < 4; ++n) {
            const int col = n0 + n * 16 + lr;
#pragma unroll
            for (int r = 0; r < 8; ++r) {
                const int row = m0 + wave * 16 + hi * 8 + r;
                out[(size_t)row * D_ + col] = (__bf16)(acc[n][r] * scale);
            }
        }
    } else {
        const int b  = m0 / S_;
        const int s0 = (m0 % S_) + wave * 16 + hi * 8;
#pragma unroll
        for (int n = 0; n < 4; ++n) {
            const int e = n0 + n * 16 + lr;
            bf16x8 v;
#pragma unroll
            for (int r = 0; r < 8; ++r) v[r] = (__bf16)acc[n][r];
            *(bf16x8*)(Vt + (size_t)b * D_ * S_ + (size_t)e * S_ + s0) = v;
        }
    }
}

// ---------------------------------------------------------------------------
// Kernels 3/5: batched bf16 GEMM with double-buffered TDM tile staging.
//   C(f32)[m,n] = sum_k A[m,k] * Bt[n,k]
// Wave 0 issues tensor_load_to_lds descriptor pairs two k-steps ahead;
// s_wait_tensorcnt(2) guarantees the current buffer has landed before the
// data-ready barrier. All 8 waves stream WMMAs from the other buffer.
// ---------------------------------------------------------------------------
__global__ __launch_bounds__(256) void gemm_bf16_tdm_kernel(
    const __bf16* __restrict__ A, const __bf16* __restrict__ Bt,
    float* __restrict__ C,
    long strideA, long strideB, long strideC,
    int lda, int ldb, int ldc, int kdim)
{
    __shared__ __align__(16) __bf16 lA[2][MB * LDSP];
    __shared__ __align__(16) __bf16 lB[2][NB * LDSP];

    A  += (size_t)blockIdx.z * strideA;
    Bt += (size_t)blockIdx.z * strideB;
    C  += (size_t)blockIdx.z * strideC;

    const int m0 = blockIdx.y * MB;
    const int n0 = blockIdx.x * NB;
    const int tid  = threadIdx.x;
    const int wave = tid >> 5;
    const int lane = tid & 31;
    const int lr   = lane & 15;
    const int hi   = lane >> 4;

    v8f acc[4] = {};
    const int steps = kdim / KB;

    if (tid < 32) {
        tdm_load_tile(A  + (size_t)m0 * lda, lA[0], MB, lda);
        tdm_load_tile(Bt + (size_t)n0 * ldb, lB[0], NB, ldb);
        if (steps > 1) {
            tdm_load_tile(A  + (size_t)m0 * lda + KB, lA[1], MB, lda);
            tdm_load_tile(Bt + (size_t)n0 * ldb + KB, lB[1], NB, ldb);
        }
    }

    for (int s = 0; s < steps; ++s) {
        const int cur = s & 1;
        if (tid < 32) {
            if (s + 1 < steps) __builtin_amdgcn_s_wait_tensorcnt(2);
            else               __builtin_amdgcn_s_wait_tensorcnt(0);
        }
        __syncthreads();   // data-ready: buffer `cur` is in LDS

        const v16bf a = load_a_frag(lA[cur] + (wave * 16 + lr) * LDSP, hi);
#pragma unroll
        for (int n = 0; n < 4; ++n) {
            const v16bf b = load_b_frag(lB[cur] + (n * 16 + lr) * LDSP, hi);
            acc[n] = __builtin_amdgcn_wmma_f32_16x16x32_bf16(
                false, a, false, b, (short)0, acc[n], false, false);
        }
        __syncthreads();   // all waves done reading `cur`; safe to overwrite

        if (tid < 32 && (s + 2) < steps) {
            const int kn = (s + 2) * KB;
            tdm_load_tile(A  + (size_t)m0 * lda + kn, lA[cur], MB, lda);
            tdm_load_tile(Bt + (size_t)n0 * ldb + kn, lB[cur], NB, ldb);
        }
    }

#pragma unroll
    for (int n = 0; n < 4; ++n) {
        const int col = n0 + n * 16 + lr;
#pragma unroll
        for (int r = 0; r < 8; ++r) {
            const int row = m0 + wave * 16 + hi * 8 + r;
            C[(size_t)row * ldc + col] = acc[n][r];
        }
    }
}

// ---------------------------------------------------------------------------
// Kernel 4: row softmax over f32 scores -> bf16 probabilities.
// ---------------------------------------------------------------------------
__global__ __launch_bounds__(256) void softmax_kernel(
    const float* __restrict__ Sc, __bf16* __restrict__ P)
{
    __shared__ float red[256];
    const size_t row = blockIdx.x;
    const float* x = Sc + row * S_;
    const int tid = threadIdx.x;

    float m = -3.4e38f;
    for (int i = tid; i < S_; i += 256) m = fmaxf(m, x[i]);
    red[tid] = m; __syncthreads();
    for (int s = 128; s > 0; s >>= 1) {
        if (tid < s) red[tid] = fmaxf(red[tid], red[tid + s]);
        __syncthreads();
    }
    m = red[0]; __syncthreads();

    float sum = 0.f;
    for (int i = tid; i < S_; i += 256) sum += __expf(x[i] - m);
    red[tid] = sum; __syncthreads();
    for (int s = 128; s > 0; s >>= 1) {
        if (tid < s) red[tid] += red[tid + s];
        __syncthreads();
    }
    const float inv = 1.0f / red[0];

    for (int i = tid; i < S_; i += 256)
        P[row * S_ + i] = (__bf16)(__expf(x[i] - m) * inv);
}

// ---------------------------------------------------------------------------
// Host-side launcher
// ---------------------------------------------------------------------------
extern "C" void kernel_launch(void* const* d_in, const int* in_sizes, int n_in,
                              void* d_out, int out_size, void* d_ws, size_t ws_size,
                              hipStream_t stream) {
    (void)in_sizes; (void)n_in; (void)out_size; (void)ws_size;

    const float* Xk = (const float*)d_in[0];   // inputs_for_keys
    const float* Xv = (const float*)d_in[1];   // inputs_for_values
    const float* Xq = (const float*)d_in[2];   // inputs_for_queries
    const float* WK = (const float*)d_in[3];
    const float* WV = (const float*)d_in[4];
    const float* WQ = (const float*)d_in[5];
    float* Out = (float*)d_out;

    char* ws = (char*)d_ws;
    const size_t WT_SZ  = (size_t)D_ * D_ * 2;          // 2 MB each
    const size_t QKV_SZ = (size_t)B_ * S_ * D_ * 2;     // 32 MB each
    const size_t SC_SZ  = (size_t)B_ * S_ * S_ * 4;     // 128 MB

    __bf16* WtQ = (__bf16*)(ws);
    __bf16* WtK = (__bf16*)(ws + WT_SZ);
    __bf16* WtV = (__bf16*)(ws + 2 * WT_SZ);
    __bf16* Qb  = (__bf16*)(ws + 3 * WT_SZ);
    __bf16* Kb  = (__bf16*)(ws + 3 * WT_SZ + QKV_SZ);
    __bf16* Vt  = (__bf16*)(ws + 3 * WT_SZ + 2 * QKV_SZ);
    float*  Sc  = (float*) (ws + 3 * WT_SZ + 3 * QKV_SZ);
    __bf16* P   = (__bf16*)(ws + 3 * WT_SZ + 3 * QKV_SZ + SC_SZ);

    // 1) W -> Wt (bf16, transposed)
    wt_convert_kernel<<<dim3(D_ / 32, D_ / 32, 3), 256, 0, stream>>>(
        WQ, WK, WV, WtQ, WtK, WtV);

    // 2) Projections: Qb (scaled), Kb, Vt
    proj_kernel<<<dim3(D_ / NB, (B_ * S_) / MB, 3), 256, 0, stream>>>(
        Xq, Xk, Xv, WtQ, WtK, WtV, Qb, Kb, Vt);

    // 3) Scores: Sc[b,q,k] = Qb[b,q,:] . Kb[b,k,:]
    gemm_bf16_tdm_kernel<<<dim3(S_ / NB, S_ / MB, B_), 256, 0, stream>>>(
        Qb, Kb, Sc,
        (long)S_ * D_, (long)S_ * D_, (long)S_ * S_,
        D_, D_, S_, D_);

    // 4) Softmax rows -> P (bf16)
    softmax_kernel<<<B_ * S_, 256, 0, stream>>>(Sc, P);

    // 5) Output: Out[b,q,e] = sum_k P[b,q,k] * V[b,k,e]  (Bt = Vt rows)
    gemm_bf16_tdm_kernel<<<dim3(D_ / NB, S_ / MB, B_), 256, 0, stream>>>(
        P, Vt, Out,
        (long)S_ * S_, (long)D_ * S_, (long)S_ * D_,
        S_, S_, D_, S_);
}
